// bahdanau_attention_13881334300822
// MI455X (gfx1250) — matrix operand order repaired and checked by reading, not compiled, not run on hardware
//
#include <hip/hip_runtime.h>
#include <hip/hip_bf16.h>
#include <stddef.h>

// ---------------------------------------------------------------------------
// Bahdanau attention, MI455X (gfx1250, wave32, WMMA).
//   B=32, S=1024, H=1024, E=2H=2048.
// Fused bf16-WMMA GEMM for tanh(context@We + be + gdec) @ Wv with atomic
// score accumulation (gamma_enc never materialized), then softmax, then a
// bandwidth-bound weighted context reduction.
// ---------------------------------------------------------------------------

#define B_N   32
#define S_LEN 1024
#define H_DIM 1024
#define E_DIM 2048

typedef __attribute__((ext_vector_type(16))) __bf16       v16bf;
typedef __attribute__((ext_vector_type(8)))  float        v8f;
typedef __attribute__((ext_vector_type(4)))  unsigned int v4u;
typedef __attribute__((ext_vector_type(2)))  float        v2f;

union FragU {            // one WMMA 16-bit operand: 8 VGPRs = 16 bf16 = 32 B
    v4u   q[2];
    v16bf v;
};

// fp32 -> packed bf16x2; prefer native v_cvt_pk_bf16_f32 when declared
__device__ __forceinline__ unsigned int pack_bf16x2(float lo, float hi) {
#if __has_builtin(__builtin_amdgcn_cvt_pk_bf16_f32)
    return __builtin_bit_cast(unsigned int,
                              __builtin_amdgcn_cvt_pk_bf16_f32(lo, hi));
#else
    unsigned int ulo = __builtin_bit_cast(unsigned int, lo);
    unsigned int uhi = __builtin_bit_cast(unsigned int, hi);
    ulo += 0x7FFFu + ((ulo >> 16) & 1u);
    uhi += 0x7FFFu + ((uhi >> 16) & 1u);
    return (ulo >> 16) | (uhi & 0xFFFF0000u);
#endif
}

// gfx1250 has a native V_TANH_F32 transcendental
__device__ __forceinline__ float fast_tanh(float x) {
#if __has_builtin(__builtin_amdgcn_tanhf)
    return __builtin_amdgcn_tanhf(x);
#elif __has_builtin(__builtin_amdgcn_tanh_f32)
    return __builtin_amdgcn_tanh_f32(x);
#else
    return tanhf(x);
#endif
}

// ---------------------------------------------------------------------------
// Kernel 1: repack We (fp32 [2048][1024], row-major) into bf16 B-fragment
// layout: [nf 0..63][kf 0..63][lane 0..31][8 dwords].
// B pack order used here (K=32): lanes 0-15 hold K=0..15 (VGPR v -> K=2v,2v+1),
// lanes 16-31 hold K=16..31; N = lane&15.
// ---------------------------------------------------------------------------
__global__ __launch_bounds__(256) void pack_we_kernel(
    const float* __restrict__ We, unsigned int* __restrict__ wpack)
{
    int t    = blockIdx.x * 256 + threadIdx.x;   // 0 .. 64*64*32-1
    int lane = t & 31;
    int kf   = (t >> 5) & 63;
    int nf   = t >> 11;
    int n     = nf * 16 + (lane & 15);
    int kbase = kf * 32 + ((lane >> 4) * 16);
    unsigned int* dst = wpack + (size_t)t * 8;
#pragma unroll
    for (int v = 0; v < 8; ++v) {
        int k = kbase + 2 * v;
        dst[v] = pack_bf16x2(We[(size_t)k * H_DIM + n],
                             We[(size_t)(k + 1) * H_DIM + n]);
    }
}

// ---------------------------------------------------------------------------
// Kernel 2: gamma_dec = dec_hidden @ Wd + bd  (32x1024, tiny) + zero scores.
// ---------------------------------------------------------------------------
__global__ __launch_bounds__(256) void gdec_zero_kernel(
    const float* __restrict__ dec_hidden, const float* __restrict__ Wd,
    const float* __restrict__ bd, float* __restrict__ gdec,
    float* __restrict__ scores)
{
    int t = blockIdx.x * 256 + threadIdx.x;      // 0..32767
    scores[t] = 0.0f;
    int b = t >> 10, h = t & (H_DIM - 1);
    const float* dh = dec_hidden + b * H_DIM;
    float acc = bd[h];
#pragma unroll 4
    for (int e = 0; e < H_DIM; ++e)
        acc += dh[e] * Wd[(size_t)e * H_DIM + h];
    gdec[t] = acc;
}

// ---------------------------------------------------------------------------
// Kernel 3: fused GEMM + tanh + dot(Wv) -> atomic scores.
// Grid: x = (B*S)/128 row-blocks, y = H/128 col-blocks. 256 thr = 8 waves.
// Waves tiled 2(M) x 4(N); each wave: 4 m-frags x 2 n-frags, K-loop of 64.
// Context fp32 staged into double-buffered LDS as bf16 in A-fragment layout
// (one s_barrier per K-step; stage k+1 overlaps WMMA of step k).
// A staged in LDS (16x32): lane half by k>=16, dword v = (k&15)>>1, same k order as B pack.
// ---------------------------------------------------------------------------
__global__ __launch_bounds__(256) void attn_gemm_scores_kernel(
    const float* __restrict__ ctx, const v4u* __restrict__ wpack,
    const float* __restrict__ be, const float* __restrict__ gdec,
    const float* __restrict__ Wv, float* __restrict__ scores)
{
    __shared__ unsigned int ldsA[2][2048];       // 2 x 8KB double buffer

    const int tid  = threadIdx.x;
    const int lane = tid & 31;
    const int wave = tid >> 5;
    const int wm   = wave & 1;                   // 0..1 along M
    const int wn   = wave >> 1;                  // 0..3 along N
    const int mblock = blockIdx.x;               // 0..255
    const int nblock = blockIdx.y;               // 0..7
    const int b  = mblock >> 3;
    const int s0 = (mblock & 7) * 128;

    // staging writer: thread -> (row m, k-group); streams 64B contiguous
    const int wr_m  = tid >> 1;                  // 0..127
    const int wr_kg = tid & 1;                   // 0..1
    const int wr_fm = wr_m >> 4;
    const int wr_mr = wr_m & 15;
    const float* arow = ctx + (size_t)(b * S_LEN + s0 + wr_m) * E_DIM;

    auto stage = [&](int kf, int buf) {
        const int kcol = kf * 32 + wr_kg * 16;
        __builtin_prefetch(arow + kcol + 64, 0, 1);   // global_prefetch_b8
        unsigned int* dst = &ldsA[buf][wr_fm * 256];
#pragma unroll
        for (int i = 0; i < 8; ++i) {
            int k = wr_kg * 16 + 2 * i;               // local K, even
            v2f c2 = *(const v2f*)(arow + kcol + 2 * i);
            unsigned int d = pack_bf16x2(c2.x, c2.y);
            int lw = wr_mr + ((k < 16) ? 0 : 16);
            int v  = (k & 15) >> 1;
            dst[lw * 8 + v] = d;
        }
    };

    v8f acc[4][2];
    const v8f vzero = {0.f, 0.f, 0.f, 0.f, 0.f, 0.f, 0.f, 0.f};
#pragma unroll
    for (int fm = 0; fm < 4; ++fm)
#pragma unroll
        for (int fn = 0; fn < 2; ++fn) acc[fm][fn] = vzero;

    stage(0, 0);
    for (int kf = 0; kf < 64; ++kf) {
        __syncthreads();                         // staging of buf(kf) visible
        if (kf + 1 < 64) stage(kf + 1, (kf + 1) & 1);
        const int buf = kf & 1;

        // B fragments straight from packed global (L2-resident, coalesced b128)
        FragU bfr[2];
#pragma unroll
        for (int fn = 0; fn < 2; ++fn) {
            int nf = nblock * 8 + wn * 2 + fn;
            const v4u* bp = wpack + ((size_t)(nf * 64 + kf) * 32 + lane) * 2;
            bfr[fn].q[0] = bp[0];
            bfr[fn].q[1] = bp[1];
        }
#pragma unroll
        for (int fm = 0; fm < 4; ++fm) {
            FragU af;
            const v4u* ap = (const v4u*)&ldsA[buf][(wm * 4 + fm) * 256 + lane * 8];
            af.q[0] = ap[0];
            af.q[1] = ap[1];
#pragma unroll
            for (int fn = 0; fn < 2; ++fn) {
                acc[fm][fn] = __builtin_amdgcn_wmma_f32_16x16x32_bf16(
                    false, af.v, false, bfr[fn].v,
                    (short)0, acc[fm][fn], false, false);
            }
        }
    }

    // Epilogue: score[b,s] += sum_n tanh(acc + be[n] + gdec[b,n]) * Wv[n]
    // C layout: VGPR r, lanes 0-15 -> (M=r, N=lane); lanes 16-31 -> (M=8+r).
    // Fold the two n-fragments before the 16-lane reduction.
    float bias[2], wv[2];
#pragma unroll
    for (int fn = 0; fn < 2; ++fn) {
        int n = nblock * 128 + wn * 32 + fn * 16 + (lane & 15);
        bias[fn] = be[n] + gdec[b * H_DIM + n];
        wv[fn]   = Wv[n];
    }
#pragma unroll
    for (int fm = 0; fm < 4; ++fm) {
#pragma unroll
        for (int r = 0; r < 8; ++r) {
            float contrib = fast_tanh(acc[fm][0][r] + bias[0]) * wv[0]
                          + fast_tanh(acc[fm][1][r] + bias[1]) * wv[1];
#pragma unroll
            for (int off = 1; off < 16; off <<= 1)
                contrib += __shfl_xor(contrib, off, 32);
            if ((lane & 15) == 0) {
                int s = s0 + wm * 64 + fm * 16 + r + ((lane >> 4) << 3);
                atomicAdd(&scores[b * S_LEN + s], contrib);
            }
        }
    }
}

// ---------------------------------------------------------------------------
// Kernel 4: softmax over S per batch. (bv shift cancels in softmax.)
// ---------------------------------------------------------------------------
__global__ __launch_bounds__(256) void softmax_kernel(
    const float* __restrict__ scores, float* __restrict__ weights)
{
    __shared__ float red[256];
    int b = blockIdx.x, tid = threadIdx.x;
    float4 v = ((const float4*)(scores + b * S_LEN))[tid];
    float m = fmaxf(fmaxf(v.x, v.y), fmaxf(v.z, v.w));
    red[tid] = m;
    __syncthreads();
    for (int o = 128; o > 0; o >>= 1) {
        if (tid < o) red[tid] = fmaxf(red[tid], red[tid + o]);
        __syncthreads();
    }
    float mx = red[0];
    __syncthreads();
    float4 e;
    e.x = __expf(v.x - mx); e.y = __expf(v.y - mx);
    e.z = __expf(v.z - mx); e.w = __expf(v.w - mx);
    red[tid] = e.x + e.y + e.z + e.w;
    __syncthreads();
    for (int o = 128; o > 0; o >>= 1) {
        if (tid < o) red[tid] += red[tid + o];
        __syncthreads();
    }
    float inv = 1.0f / red[0];
    float4 w4;
    w4.x = e.x * inv; w4.y = e.y * inv; w4.z = e.z * inv; w4.w = e.w * inv;
    ((float4*)(weights + b * S_LEN))[tid] = w4;
}

// ---------------------------------------------------------------------------
// Kernel 5: c_t[b,e] = sum_s weights[b,s] * context[b,s,e]. Pure bandwidth.
// ---------------------------------------------------------------------------
__global__ __launch_bounds__(256) void context_reduce_kernel(
    const float* __restrict__ ctx, const float* __restrict__ weights,
    float* __restrict__ out)
{
    __shared__ float wsh[S_LEN];
    int b = blockIdx.y;
    int e = blockIdx.x * 256 + threadIdx.x;
    for (int i = threadIdx.x; i < S_LEN; i += 256)
        wsh[i] = weights[b * S_LEN + i];
    __syncthreads();
    const float* p = ctx + (size_t)b * S_LEN * E_DIM + e;
    float acc = 0.0f;
#pragma unroll 4
    for (int s = 0; s < S_LEN; ++s)
        acc += wsh[s] * p[(size_t)s * E_DIM];
    out[b * E_DIM + e] = acc;
}

// ---------------------------------------------------------------------------
extern "C" void kernel_launch(void* const* d_in, const int* in_sizes, int n_in,
                              void* d_out, int out_size, void* d_ws, size_t ws_size,
                              hipStream_t stream)
{
    const float* ctx = (const float*)d_in[0];   // [32,1024,2048]
    const float* dec = (const float*)d_in[1];   // [32,1024]
    const float* We  = (const float*)d_in[2];   // [2048,1024]
    const float* be  = (const float*)d_in[3];   // [1024]
    const float* Wd  = (const float*)d_in[4];   // [1024,1024]
    const float* bd  = (const float*)d_in[5];   // [1024]
    const float* Wv  = (const float*)d_in[6];   // [1024,1] -> [1024]
    (void)d_in[7];                              // bv cancels in softmax

    float* c_t     = (float*)d_out;             // [32,2048]
    float* weights = c_t + B_N * E_DIM;         // [32,1024]

    float* gdec   = (float*)d_ws;                        // 32*1024 f32
    float* scores = gdec + B_N * H_DIM;                  // 32*1024 f32
    unsigned int* wpack = (unsigned int*)(scores + B_N * S_LEN); // 4 MB bf16 pack

    pack_we_kernel<<<512, 256, 0, stream>>>(We, wpack);
    gdec_zero_kernel<<<128, 256, 0, stream>>>(dec, Wd, bd, gdec, scores);
    attn_gemm_scores_kernel<<<dim3(256, 8), 256, 0, stream>>>(
        ctx, (const v4u*)wpack, be, gdec, Wv, scores);
    softmax_kernel<<<B_N, 256, 0, stream>>>(scores, weights);
    context_reduce_kernel<<<dim3(E_DIM / 256, B_N), 256, 0, stream>>>(
        ctx, weights, c_t);
}
